// Block_82291573391930
// MI455X (gfx1250) — compile-verified, matrix operand order checked by reading, not run on hardware
//
#include <hip/hip_runtime.h>
#include <math.h>

typedef _Float16 v16h __attribute__((ext_vector_type(16)));
typedef _Float16 v8h  __attribute__((ext_vector_type(8)));
typedef _Float16 v4h  __attribute__((ext_vector_type(4)));
typedef float    v8f  __attribute__((ext_vector_type(8)));
typedef float    v4f  __attribute__((ext_vector_type(4)));

#define LN_EPS 1e-5f
static constexpr float BN_RS_C = 0.9999950000374997f; // 1/sqrt(1+1e-5)

// ---------- sizes ----------
#define NGROUP 6272          // B*NPATCH
#define BDIM   32
#define NPATCH 196
#define NTOK   197
#define DIM    640
#define IN_DIM 40
#define NPIX   16
#define HID    2560
#define CA_INNER 256

static constexpr size_t SZ_PIX   = (size_t)NGROUP * NPIX * IN_DIM;   // 4,014,080
static constexpr size_t SZ_PATCH = (size_t)BDIM * NTOK * DIM;        // 4,034,560
static constexpr size_t SZ_QK    = (size_t)BDIM * NTOK * 2 * DIM;    // 8,069,120
static constexpr size_t SZ_CKV   = (size_t)BDIM * NTOK * CA_INNER;   // 1,613,824
static constexpr size_t SZ_C1    = (size_t)NGROUP * HID;             // 16,056,320
static constexpr size_t SZ_PT    = (size_t)NGROUP * DIM;             // 4,014,080
static constexpr size_t SZ_SE    = (size_t)BDIM * HID;               // 81,920

// ---------- helpers ----------
__device__ __forceinline__ float wave_sum_f(float v) {
#pragma unroll
  for (int off = 16; off > 0; off >>= 1) v += __shfl_xor(v, off, 32);
  return v;
}
__device__ __forceinline__ float hswish_f(float v) {
  float t = fminf(fmaxf(v + 3.f, 0.f), 6.f);
  return v * t * (1.f / 6.f);
}
__device__ __forceinline__ float silu_f(float v) { return v / (1.f + __expf(-v)); }

// =====================================================================
// Generic WMMA GEMM: Out[m,n] = act( (sum_k A[m,k]*W[n,k]) * scale[n]*smul
//                                    + bias[n] + res[m,n] )
// A row-major [M,K], W row-major [N,K], Out row-major [M,N].
// Block = 128 threads (4 waves); tile 64x64, K-step 32, f16 WMMA,
// double-buffered LDS, one barrier per K-step, global loads overlap WMMA.
// =====================================================================
template<int ACT>
__global__ void __launch_bounds__(128)
wmma_gemm(const float* __restrict__ A, const float* __restrict__ W,
          const float* __restrict__ scale, float smul,
          const float* __restrict__ bias, const float* __restrict__ res,
          float* __restrict__ Out, int M, int N, int K)
{
  __shared__ _Float16 As[2][64][32];
  __shared__ _Float16 Bs[2][64][32];
  const int tid = threadIdx.x, wave = tid >> 5, lane = tid & 31;
  const int m0 = blockIdx.y * 64, n0 = blockIdx.x * 64;
  const bool full = (m0 + 64 <= M) && (n0 + 64 <= N);   // block-uniform
  const int rr = tid >> 3;                // base row 0..15 (stride 16 over 4 chunks)
  const int c4 = (tid & 7) << 2;          // float4 column within the 32-wide K slab
  v8f acc[4] = {};
  const int ml = lane & 15;
  const int kb = (lane < 16) ? 0 : 8;     // K base per ISA A/B fragment layout

  v4f av[4], bv[4];

  auto loadTiles = [&](int k0) {
    if (full) {
#pragma unroll
      for (int i = 0; i < 4; ++i) {
        int r = rr + i * 16;
        av[i] = *(const v4f*)&A[(size_t)(m0 + r) * K + k0 + c4];
        bv[i] = *(const v4f*)&W[(size_t)(n0 + r) * K + k0 + c4];
      }
      if (k0 + 32 < K) {                  // global_prefetch_b8 for tile after next
        __builtin_prefetch(&A[(size_t)(m0 + rr) * K + k0 + 32], 0, 1);
        __builtin_prefetch(&W[(size_t)(n0 + rr) * K + k0 + 32], 0, 1);
      }
    } else {
#pragma unroll
      for (int i = 0; i < 4; ++i) {
        int r = rr + i * 16;
        int mg = m0 + r, ng = n0 + r;
        v4f a = {0.f, 0.f, 0.f, 0.f}, b = {0.f, 0.f, 0.f, 0.f};
        if (mg < M) a = *(const v4f*)&A[(size_t)mg * K + k0 + c4];
        if (ng < N) b = *(const v4f*)&W[(size_t)ng * K + k0 + c4];
        av[i] = a; bv[i] = b;
      }
    }
  };
  auto storeTiles = [&](int buf) {
#pragma unroll
    for (int i = 0; i < 4; ++i) {
      int r = rr + i * 16;
      v4h ha, hb;
#pragma unroll
      for (int e = 0; e < 4; ++e) { ha[e] = (_Float16)av[i][e]; hb[e] = (_Float16)bv[i][e]; }
      *(v4h*)&As[buf][r][c4] = ha;
      *(v4h*)&Bs[buf][r][c4] = hb;
    }
  };

  const int T = K >> 5;                   // all K used are multiples of 32
  loadTiles(0);
  storeTiles(0);
  __syncthreads();

  for (int i = 0; i < T; ++i) {
    const int buf = i & 1;
    if (i + 1 < T) loadTiles((i + 1) << 5);    // overlaps with WMMA below
    {
      v8h alo = *(const v8h*)&As[buf][(wave << 4) + ml][kb];
      v8h ahi = *(const v8h*)&As[buf][(wave << 4) + ml][kb + 16];
      v16h af = __builtin_shufflevector(alo, ahi, 0,1,2,3,4,5,6,7,8,9,10,11,12,13,14,15);
#pragma unroll
      for (int j = 0; j < 4; ++j) {
        v8h blo = *(const v8h*)&Bs[buf][(j << 4) + ml][kb];
        v8h bhi = *(const v8h*)&Bs[buf][(j << 4) + ml][kb + 16];
        v16h bf = __builtin_shufflevector(blo, bhi, 0,1,2,3,4,5,6,7,8,9,10,11,12,13,14,15);
        acc[j] = __builtin_amdgcn_wmma_f32_16x16x32_f16(
            false, af, false, bf, (short)0, acc[j], false, false);
      }
    }
    if (i + 1 < T) storeTiles(buf ^ 1);
    __syncthreads();
  }

  const int mb = m0 + (wave << 4) + ((lane < 16) ? 0 : 8);
#pragma unroll
  for (int j = 0; j < 4; ++j) {
    int n = n0 + (j << 4) + ml;
    if (n >= N) continue;
    float sc = scale ? scale[n] * smul : 1.f;
    float bi = bias ? bias[n] : 0.f;
#pragma unroll
    for (int r = 0; r < 8; ++r) {
      int mm = mb + r;
      if (mm >= M) continue;
      float v = acc[j][r] * sc + bi;
      if (res) v += res[(size_t)mm * N + n];
      if (ACT == 1) v = hswish_f(v);
      Out[(size_t)mm * N + n] = v;
    }
  }
}

// =====================================================================
// LayerNorm over rows (one block of 64 threads per row)
// =====================================================================
__global__ void __launch_bounds__(64)
ln_rows_kernel(const float* __restrict__ X, const float* __restrict__ g,
               const float* __restrict__ b, float* __restrict__ Y, int C)
{
  __shared__ float red[64];
  const int row = blockIdx.x, tid = threadIdx.x;
  const float* x = X + (size_t)row * C;
  float s = 0.f;
  for (int c = tid; c < C; c += 64) s += x[c];
  red[tid] = s; __syncthreads();
  for (int off = 32; off > 0; off >>= 1) { if (tid < off) red[tid] += red[tid + off]; __syncthreads(); }
  float mu = red[0] / (float)C;
  __syncthreads();
  float v = 0.f;
  for (int c = tid; c < C; c += 64) { float d = x[c] - mu; v += d * d; }
  red[tid] = v; __syncthreads();
  for (int off = 32; off > 0; off >>= 1) { if (tid < off) red[tid] += red[tid + off]; __syncthreads(); }
  float rs = rsqrtf(red[0] / (float)C + LN_EPS);
  float* y = Y + (size_t)row * C;
  for (int c = tid; c < C; c += 64) y[c] = (x[c] - mu) * rs * g[c] + b[c];
}

// =====================================================================
// Inner path: one 64-thread block per pixel group (16 x 40)
// =====================================================================
__device__ __forceinline__ void group_ln(const float* base, const float* g,
                                         const float* b, float xln[16][40], int tid)
{
  if (tid < 16) {
    float mu = 0.f;
    for (int c = 0; c < 40; ++c) mu += base[tid * 40 + c];
    mu *= (1.f / 40.f);
    float var = 0.f;
    for (int c = 0; c < 40; ++c) { float d = base[tid * 40 + c] - mu; var += d * d; }
    var *= (1.f / 40.f);
    float rs = rsqrtf(var + LN_EPS);
    for (int c = 0; c < 40; ++c)
      xln[tid][c] = (base[tid * 40 + c] - mu) * rs * g[c] + b[c];
  }
}

__global__ void __launch_bounds__(64)
inner_mamba_kernel(float* __restrict__ P,
                   const float* lng, const float* lnb,
                   const float* in_proj_w, const float* conv_x_w, const float* conv_z_w,
                   const float* x_proj_w, const float* dt_proj_w, const float* dt_proj_b,
                   const float* A_log, const float* Dp, const float* out_proj_w)
{
  __shared__ float xln[16][40];
  __shared__ float xm[40][16], zz[40][16];
  __shared__ float xmc[40][16], zc[40][16];
  __shared__ float xdbl[16][36];
  __shared__ float ycat[16][80];
  const int g = blockIdx.x, tid = threadIdx.x;
  float* base = P + (size_t)g * 640;

  group_ln(base, lng, lnb, xln, tid);
  __syncthreads();

  for (int idx = tid; idx < 1280; idx += 64) {           // in_proj 16x80
    int l = idx / 80, j = idx % 80;
    float v = 0.f; const float* w = in_proj_w + j * 40;
    for (int c = 0; c < 40; ++c) v += xln[l][c] * w[c];
    if (j < 40) xm[j][l] = v; else zz[j - 40][l] = v;
  }
  __syncthreads();

  for (int idx = tid; idx < 640; idx += 64) {            // dwconv1d + silu
    int d = idx / 16, l = idx % 16;
    float a = 0.f, bz = 0.f;
    for (int k = 0; k < 4; ++k) {
      int ls = l + k - 1;
      if (ls >= 0 && ls < 16) {
        a  += conv_x_w[d * 4 + k] * xm[d][ls];
        bz += conv_z_w[d * 4 + k] * zz[d][ls];
      }
    }
    xmc[d][l] = silu_f(a);
    zc[d][l]  = silu_f(bz);
  }
  __syncthreads();

  for (int idx = tid; idx < 560; idx += 64) {            // x_dbl 16x35
    int l = idx / 35, e = idx % 35;
    float v = 0.f; const float* w = x_proj_w + e * 40;
    for (int d = 0; d < 40; ++d) v += xmc[d][l] * w[d];
    xdbl[l][e] = v;
  }
  for (int idx = tid; idx < 640; idx += 64) {            // z half of ycat
    int l = idx / 40, d = idx % 40;
    ycat[l][40 + d] = zc[d][l];
  }
  __syncthreads();

  for (int d = tid; d < 40; d += 64) {                   // selective scan
    float Ad[16], h[16];
#pragma unroll
    for (int n = 0; n < 16; ++n) { Ad[n] = -__expf(A_log[d * 16 + n]); h[n] = 0.f; }
    float w0 = dt_proj_w[d * 3], w1 = dt_proj_w[d * 3 + 1], w2 = dt_proj_w[d * 3 + 2];
    float bb = dt_proj_b[d], Dd = Dp[d];
    for (int l = 0; l < 16; ++l) {
      float dt = xdbl[l][0] * w0 + xdbl[l][1] * w1 + xdbl[l][2] * w2 + bb;
      float delta = (dt > 20.f) ? dt : log1pf(__expf(dt));
      float u = xmc[d][l];
      float y = 0.f;
#pragma unroll
      for (int n = 0; n < 16; ++n) {
        float dA = __expf(delta * Ad[n]);
        float dBu = delta * xdbl[l][3 + n] * u;
        h[n] = dA * h[n] + dBu;
        y += h[n] * xdbl[l][19 + n];
      }
      ycat[l][d] = y + u * Dd;
    }
  }
  __syncthreads();

  for (int idx = tid; idx < 640; idx += 64) {            // out_proj + residual
    int l = idx / 40, c = idx % 40;
    float v = 0.f; const float* w = out_proj_w + c * 80;
    for (int j = 0; j < 80; ++j) v += ycat[l][j] * w[j];
    base[l * 40 + c] += v;
  }
}

__global__ void __launch_bounds__(64)
inner_attn_kernel(float* __restrict__ P,
                  const float* lng, const float* lnb,
                  const float* qk_w, const float* v_w,
                  const float* proj_w, const float* proj_b)
{
  __shared__ float xln[16][40];
  __shared__ float qk[16][80];
  __shared__ float vv[16][40];
  __shared__ float ao[16][40];
  const int g = blockIdx.x, tid = threadIdx.x;
  float* base = P + (size_t)g * 640;

  group_ln(base, lng, lnb, xln, tid);
  __syncthreads();

  for (int idx = tid; idx < 1280; idx += 64) {
    int l = idx / 80, j = idx % 80;
    float v = 0.f; const float* w = qk_w + j * 40;
    for (int c = 0; c < 40; ++c) v += xln[l][c] * w[c];
    qk[l][j] = v;
  }
  for (int idx = tid; idx < 640; idx += 64) {
    int l = idx / 40, j = idx % 40;
    float v = 0.f; const float* w = v_w + j * 40;
    for (int c = 0; c < 40; ++c) v += xln[l][c] * w[c];
    vv[l][j] = v;
  }
  __syncthreads();

  {                                                       // 4 heads x 16 rows
    int h = tid / 16, i = tid % 16;                       // tid in [0,64)
    float s[16]; float mx = -1e30f;
    const float scl = rsqrtf(10.f);
#pragma unroll
    for (int j = 0; j < 16; ++j) {
      float v = 0.f;
      for (int e = 0; e < 10; ++e) v += qk[i][h * 10 + e] * qk[j][40 + h * 10 + e];
      s[j] = v * scl; mx = fmaxf(mx, s[j]);
    }
    float den = 0.f;
#pragma unroll
    for (int j = 0; j < 16; ++j) { s[j] = __expf(s[j] - mx); den += s[j]; }
    float inv = 1.f / den;
    for (int e = 0; e < 10; ++e) {
      float o = 0.f;
#pragma unroll
      for (int j = 0; j < 16; ++j) o += s[j] * vv[j][h * 10 + e];
      ao[i][h * 10 + e] = o * inv;
    }
  }
  __syncthreads();

  for (int idx = tid; idx < 640; idx += 64) {
    int l = idx / 40, c = idx % 40;
    float v = proj_b[c]; const float* w = proj_w + c * 40;
    for (int k = 0; k < 40; ++k) v += ao[l][k] * w[k];
    base[l * 40 + c] += v;
  }
}

__global__ void __launch_bounds__(64)
inner_mlp_kernel(float* __restrict__ P,
                 const float* lng, const float* lnb,
                 const float* fc1_w, const float* fc1_b,
                 const float* fc2_w, const float* fc2_b)
{
  __shared__ float xln[16][40];
  __shared__ float h1[16][160];
  const int g = blockIdx.x, tid = threadIdx.x;
  float* base = P + (size_t)g * 640;

  group_ln(base, lng, lnb, xln, tid);
  __syncthreads();

  for (int idx = tid; idx < 2560; idx += 64) {
    int l = idx / 160, j = idx % 160;
    float v = fc1_b[j]; const float* w = fc1_w + j * 40;
    for (int c = 0; c < 40; ++c) v += xln[l][c] * w[c];
    h1[l][j] = 0.5f * v * (1.f + erff(v * 0.70710678f));   // exact gelu
  }
  __syncthreads();

  for (int idx = tid; idx < 640; idx += 64) {
    int l = idx / 40, c = idx % 40;
    float v = fc2_b[c]; const float* w = fc2_w + c * 160;
    for (int j = 0; j < 160; ++j) v += h1[l][j] * w[j];
    base[l * 40 + c] += v;
  }
}

// =====================================================================
// Outer attention: one block per (b,h, 8 query rows); wave per row.
// Writes softmax probs to Wout (output #3) and context into CTX.
// =====================================================================
__global__ void __launch_bounds__(256)
outer_attn_kernel(const float* __restrict__ QK, const float* __restrict__ Vv,
                  float* __restrict__ Wout, float* __restrict__ CTX)
{
  __shared__ _Float16 Ks[197][64];
  __shared__ _Float16 Vs[197][64];
  __shared__ float srow[8][200];
  const int bh = blockIdx.x, b = bh / 10, h = bh % 10;
  const int tid = threadIdx.x, wave = tid >> 5, lane = tid & 31;

  for (int idx = tid; idx < 197 * 64; idx += 256) {
    int j = idx >> 6, d = idx & 63;
    Ks[j][d] = (_Float16)QK[((size_t)(b * 197 + j)) * 1280 + 640 + h * 64 + d];
    Vs[j][d] = (_Float16)Vv[((size_t)(b * 197 + j)) * 640 + h * 64 + d];
  }
  __syncthreads();

  const int i = blockIdx.y * 8 + wave;
  const bool valid = (i < 197);
  const int ic = valid ? i : 196;
  const float q0 = QK[((size_t)(b * 197 + ic)) * 1280 + h * 64 + 2 * lane];
  const float q1 = QK[((size_t)(b * 197 + ic)) * 1280 + h * 64 + 2 * lane + 1];
  const float scl = 0.125f;                               // 64^-0.5
  float mx = -1e30f;
  for (int j = 0; j < 197; ++j) {
    float p = q0 * (float)Ks[j][2 * lane] + q1 * (float)Ks[j][2 * lane + 1];
    float s = wave_sum_f(p) * scl;
    srow[wave][j] = s;                                    // same value, all lanes
    mx = fmaxf(mx, s);
  }
  float den = 0.f;
  for (int j = lane; j < 197; j += 32) {
    float e = __expf(srow[wave][j] - mx);
    srow[wave][j] = e;
    den += e;
  }
  den = wave_sum_f(den);
  float inv = 1.f / den;
  for (int j = lane; j < 197; j += 32) {
    float pv = srow[wave][j] * inv;
    srow[wave][j] = pv;
    if (valid) Wout[(((size_t)(b * 10 + h) * 197 + i) * 197) + j] = pv;
  }
  float c0 = 0.f, c1 = 0.f;
  for (int j = 0; j < 197; ++j) {
    float pv = srow[wave][j];
    c0 += pv * (float)Vs[j][2 * lane];
    c1 += pv * (float)Vs[j][2 * lane + 1];
  }
  if (valid) {
    CTX[((size_t)(b * 197 + i)) * 640 + h * 64 + 2 * lane]     = c0;
    CTX[((size_t)(b * 197 + i)) * 640 + h * 64 + 2 * lane + 1] = c1;
  }
}

// =====================================================================
// Cross attention (flash-style, heads looped, q-proj and out-proj fused).
// Writes the FINAL pixel output directly.
// =====================================================================
__global__ void __launch_bounds__(256)
cross_kernel(const float* __restrict__ P, const float* __restrict__ CK,
             const float* __restrict__ CV,
             const float* __restrict__ to_q_w, const float* __restrict__ to_out_w,
             const float* __restrict__ to_out_b, float* __restrict__ OutPix)
{
  __shared__ _Float16 Kh[197][64];
  __shared__ _Float16 Vh[197][64];
  __shared__ float prow[8][40];
  __shared__ float ctxb[8][64];
  const int b = blockIdx.x;
  const int tid = threadIdx.x, wave = tid >> 5, lane = tid & 31;
  const int qi = blockIdx.y * 8 + wave;                   // 0..3135 exactly

  const float* pr = P + ((size_t)b * 3136 + qi) * 40;
  for (int c = lane; c < 40; c += 32) prow[wave][c] = pr[c];

  float o0 = 0.f, o1 = 0.f;                               // out dims lane, lane+32
  for (int h = 0; h < 4; ++h) {
    __syncthreads();
    for (int idx = tid; idx < 197 * 64; idx += 256) {
      int j = idx >> 6, d = idx & 63;
      Kh[j][d] = (_Float16)CK[((size_t)(b * 197 + j)) * 256 + h * 64 + d];
      Vh[j][d] = (_Float16)CV[((size_t)(b * 197 + j)) * 256 + h * 64 + d];
    }
    __syncthreads();

    float q0 = 0.f, q1 = 0.f;                             // fused q projection
    for (int c = 0; c < 40; ++c) {
      float pv = prow[wave][c];
      q0 += pv * to_q_w[(h * 64 + 2 * lane) * 40 + c];
      q1 += pv * to_q_w[(h * 64 + 2 * lane + 1) * 40 + c];
    }
    const float scl = 0.125f;
    float mx = -1e30f, den = 0.f, c0 = 0.f, c1 = 0.f;
    for (int j = 0; j < 197; ++j) {                       // online softmax
      float p = q0 * (float)Kh[j][2 * lane] + q1 * (float)Kh[j][2 * lane + 1];
      float s = wave_sum_f(p) * scl;
      float mn = fmaxf(mx, s);
      float corr = __expf(mx - mn);
      float e = __expf(s - mn);
      den = den * corr + e;
      c0 = c0 * corr + e * (float)Vh[j][2 * lane];
      c1 = c1 * corr + e * (float)Vh[j][2 * lane + 1];
      mx = mn;
    }
    float inv = 1.f / den;
    ctxb[wave][2 * lane]     = c0 * inv;
    ctxb[wave][2 * lane + 1] = c1 * inv;
    for (int e = 0; e < 64; ++e) {                        // fused out-proj
      float cv = ctxb[wave][e];
      o0 += cv * to_out_w[lane * 256 + h * 64 + e];
      if (lane < 8) o1 += cv * to_out_w[(lane + 32) * 256 + h * 64 + e];
    }
  }
  float* op = OutPix + ((size_t)b * 3136 + qi) * 40;
  op[lane] = o0 + to_out_b[lane];
  if (lane < 8) op[lane + 32] = o1 + to_out_b[lane + 32];
}

// =====================================================================
// Locality-FF helpers + glue kernels
// =====================================================================
__global__ void __launch_bounds__(256)
add_patch_kernel(float* __restrict__ Qp, const float* __restrict__ T)
{
  int idx = blockIdx.x * 256 + threadIdx.x;               // NGROUP*DIM
  if (idx >= NGROUP * DIM) return;
  int m = idx / DIM, c = idx % DIM;
  int b = m / NPATCH, t = m % NPATCH;
  Qp[((size_t)(b * NTOK + 1 + t)) * DIM + c] += T[idx];
}

__global__ void __launch_bounds__(256)
extract_pt_kernel(const float* __restrict__ Qp, float* __restrict__ PT)
{
  int idx = blockIdx.x * 256 + threadIdx.x;
  if (idx >= NGROUP * DIM) return;
  int m = idx / DIM, c = idx % DIM;
  int b = m / NPATCH, t = m % NPATCH;
  PT[idx] = Qp[((size_t)(b * NTOK + 1 + t)) * DIM + c];
}

__global__ void __launch_bounds__(256)
dwconv_kernel(const float* __restrict__ C1, const float* __restrict__ dw_w,
              const float* __restrict__ bn2g, const float* __restrict__ bn2b,
              float* __restrict__ DW)
{
  size_t idx = (size_t)blockIdx.x * 256 + threadIdx.x;    // NGROUP*HID
  if (idx >= (size_t)NGROUP * HID) return;
  int c = (int)(idx % HID);
  int m = (int)(idx / HID);
  int b = m / NPATCH, t = m % NPATCH, y = t / 14, x = t % 14;
  float acc = 0.f;
#pragma unroll
  for (int dy = 0; dy < 3; ++dy) {
    int yy = y + dy - 1; if (yy < 0 || yy >= 14) continue;
#pragma unroll
    for (int dx = 0; dx < 3; ++dx) {
      int xx = x + dx - 1; if (xx < 0 || xx >= 14) continue;
      acc += C1[((size_t)(b * NPATCH + yy * 14 + xx)) * HID + c] * dw_w[c * 9 + dy * 3 + dx];
    }
  }
  float v = acc * (bn2g[c] * BN_RS_C) + bn2b[c];
  DW[idx] = hswish_f(v);
}

__global__ void __launch_bounds__(256)
semean_kernel(const float* __restrict__ DW, float* __restrict__ SM)
{
  int idx = blockIdx.x * 256 + threadIdx.x;               // BDIM*HID
  if (idx >= BDIM * HID) return;
  int b = idx / HID, c = idx % HID;
  float s = 0.f;
  const float* p = DW + (size_t)b * NPATCH * HID + c;
  for (int t = 0; t < NPATCH; ++t) s += p[(size_t)t * HID];
  SM[idx] = s * (1.f / (float)NPATCH);
}

__global__ void __launch_bounds__(256)
se_kernel(const float* __restrict__ SM, const float* __restrict__ w1,
          const float* __restrict__ b1, const float* __restrict__ w2,
          const float* __restrict__ b2, float* __restrict__ SS)
{
  __shared__ float red[256];
  __shared__ float s4[4];
  const int b = blockIdx.x, tid = threadIdx.x;
  for (int j = 0; j < 4; ++j) {
    float p = 0.f;
    for (int c = tid; c < HID; c += 256) p += SM[b * HID + c] * w1[j * HID + c];
    red[tid] = p; __syncthreads();
    for (int off = 128; off > 0; off >>= 1) { if (tid < off) red[tid] += red[tid + off]; __syncthreads(); }
    if (tid == 0) s4[j] = fmaxf(red[0] + b1[j], 0.f);
    __syncthreads();
  }
  for (int c = tid; c < HID; c += 256) {
    float v = s4[0] * w2[c * 4] + s4[1] * w2[c * 4 + 1]
            + s4[2] * w2[c * 4 + 2] + s4[3] * w2[c * 4 + 3] + b2[c];
    SS[b * HID + c] = fminf(fmaxf(v + 3.f, 0.f), 6.f) * (1.f / 6.f);
  }
}

__global__ void __launch_bounds__(256)
sescale_kernel(float* __restrict__ DW, const float* __restrict__ SS)
{
  size_t idx = (size_t)blockIdx.x * 256 + threadIdx.x;    // NGROUP*HID
  if (idx >= (size_t)NGROUP * HID) return;
  int c = (int)(idx % HID);
  int b = (int)(idx / HID) / NPATCH;
  DW[idx] *= SS[b * HID + c];
}

__global__ void __launch_bounds__(256)
writeback_kernel(const float* __restrict__ Qp, const float* __restrict__ C2,
                 float* __restrict__ OutPatch)
{
  int idx = blockIdx.x * 256 + threadIdx.x;               // BDIM*NTOK*DIM
  if (idx >= BDIM * NTOK * DIM) return;
  int row = idx / DIM, c = idx % DIM;
  int b = row / NTOK, t = row % NTOK;
  float v = Qp[idx];
  if (t > 0) v += C2[((size_t)(b * NPATCH + t - 1)) * DIM + c];
  OutPatch[idx] = v;
}

// =====================================================================
// Launcher
// =====================================================================
extern "C" void kernel_launch(void* const* d_in, const int* in_sizes, int n_in,
                              void* d_out, int out_size, void* d_ws, size_t ws_size,
                              hipStream_t stream)
{
  (void)in_sizes; (void)n_in; (void)out_size; (void)ws_size;
#define DIN(i) ((const float*)d_in[(i)])
  // flattened input order (depth-first over setup_inputs dict)
  // 0 pixel, 1 patch, 2-3 ln_in_mamba, 4-5 ln_in, 6-7 ln_mlp_in, 8-9 ln_proj,
  // 10-11 ln_out, 12-20 mamba, 21-24 attn_in, 25-28 mlp_in, 29 proj_w, 30 proj_b,
  // 31-34 attn_out, 35-39 cross, 40-52 conv

  float* ws = (float*)d_ws;
  float* P        = ws; ws += SZ_PIX;
  float* Qp       = ws; ws += SZ_PATCH;
  float* LNB      = ws; ws += SZ_PATCH;      // reused for pixel-LN (smaller) then patch-LN
  float* PROJ_TMP = ws; ws += SZ_PT;
  float* QK       = ws; ws += SZ_QK;
  float* Vv       = ws; ws += SZ_PATCH;
  float* CTX      = ws; ws += SZ_PATCH;
  float* CK       = ws; ws += SZ_CKV;
  float* CV       = ws; ws += SZ_CKV;
  float* PT       = ws; ws += SZ_PT;
  float* C1       = ws; ws += SZ_C1;
  float* DW       = ws; ws += SZ_C1;
  float* SMEAN    = ws; ws += SZ_SE;
  float* SSCALE   = ws; ws += SZ_SE;
  float* C2       = ws; ws += SZ_PT;

  float* out_pix   = (float*)d_out;
  float* out_patch = out_pix + SZ_PIX;
  float* out_w     = out_patch + SZ_PATCH;

  hipMemcpyAsync(P,  d_in[0], SZ_PIX   * sizeof(float), hipMemcpyDeviceToDevice, stream);
  hipMemcpyAsync(Qp, d_in[1], SZ_PATCH * sizeof(float), hipMemcpyDeviceToDevice, stream);

  // ---- inner pixel path ----
  inner_mamba_kernel<<<NGROUP, 64, 0, stream>>>(P, DIN(2), DIN(3), DIN(12), DIN(13),
      DIN(14), DIN(15), DIN(16), DIN(17), DIN(18), DIN(19), DIN(20));
  inner_attn_kernel<<<NGROUP, 64, 0, stream>>>(P, DIN(4), DIN(5), DIN(21), DIN(22),
      DIN(23), DIN(24));
  inner_mlp_kernel<<<NGROUP, 64, 0, stream>>>(P, DIN(6), DIN(7), DIN(25), DIN(26),
      DIN(27), DIN(28));

  // ---- pixel -> patch projection ----
  ln_rows_kernel<<<NGROUP * NPIX, 64, 0, stream>>>(P, DIN(8), DIN(9), LNB, IN_DIM);
  wmma_gemm<0><<<dim3(DIM / 64, (NGROUP + 63) / 64), 128, 0, stream>>>(
      LNB, DIN(29), nullptr, 1.f, DIN(30), nullptr, PROJ_TMP, NGROUP, DIM, DIM);
  add_patch_kernel<<<(NGROUP * DIM + 255) / 256, 256, 0, stream>>>(Qp, PROJ_TMP);

  // ---- outer attention ----
  ln_rows_kernel<<<BDIM * NTOK, 64, 0, stream>>>(Qp, DIN(10), DIN(11), LNB, DIM);
  wmma_gemm<0><<<dim3(2 * DIM / 64, (BDIM * NTOK + 63) / 64), 128, 0, stream>>>(
      LNB, DIN(31), nullptr, 1.f, nullptr, nullptr, QK, BDIM * NTOK, 2 * DIM, DIM);
  wmma_gemm<0><<<dim3(DIM / 64, (BDIM * NTOK + 63) / 64), 128, 0, stream>>>(
      LNB, DIN(32), nullptr, 1.f, nullptr, nullptr, Vv, BDIM * NTOK, DIM, DIM);
  outer_attn_kernel<<<dim3(BDIM * 10, (NTOK + 7) / 8), 256, 0, stream>>>(QK, Vv, out_w, CTX);
  wmma_gemm<0><<<dim3(DIM / 64, (BDIM * NTOK + 63) / 64), 128, 0, stream>>>(
      CTX, DIN(33), nullptr, 1.f, DIN(34), Qp, Qp, BDIM * NTOK, DIM, DIM);

  // ---- cross attention (final pixel output) ----
  wmma_gemm<0><<<dim3(CA_INNER / 64, (BDIM * NTOK + 63) / 64), 128, 0, stream>>>(
      Qp, DIN(36), nullptr, 1.f, nullptr, nullptr, CK, BDIM * NTOK, CA_INNER, DIM);
  wmma_gemm<0><<<dim3(CA_INNER / 64, (BDIM * NTOK + 63) / 64), 128, 0, stream>>>(
      Qp, DIN(37), nullptr, 1.f, nullptr, nullptr, CV, BDIM * NTOK, CA_INNER, DIM);
  cross_kernel<<<dim3(BDIM, 3136 / 8), 256, 0, stream>>>(P, CK, CV, DIN(35), DIN(38),
      DIN(39), out_pix);

  // ---- locality feed-forward (final patch output) ----
  extract_pt_kernel<<<(NGROUP * DIM + 255) / 256, 256, 0, stream>>>(Qp, PT);
  wmma_gemm<1><<<dim3(HID / 64, NGROUP / 64), 128, 0, stream>>>(
      PT, DIN(40), DIN(41), BN_RS_C, DIN(42), nullptr, C1, NGROUP, HID, DIM);
  dwconv_kernel<<<(int)(((size_t)NGROUP * HID + 255) / 256), 256, 0, stream>>>(
      C1, DIN(43), DIN(44), DIN(45), DW);
  semean_kernel<<<(BDIM * HID + 255) / 256, 256, 0, stream>>>(DW, SMEAN);
  se_kernel<<<BDIM, 256, 0, stream>>>(SMEAN, DIN(46), DIN(47), DIN(48), DIN(49), SSCALE);
  sescale_kernel<<<(int)(((size_t)NGROUP * HID + 255) / 256), 256, 0, stream>>>(DW, SSCALE);
  wmma_gemm<0><<<dim3(DIM / 64, NGROUP / 64), 128, 0, stream>>>(
      DW, DIN(50), DIN(51), BN_RS_C, DIN(52), nullptr, C2, NGROUP, DIM, HID);
  writeback_kernel<<<(BDIM * NTOK * DIM + 255) / 256, 256, 0, stream>>>(Qp, C2, out_patch);
#undef DIN
}
